// SupConLoss_57114475102496
// MI455X (gfx1250) — matrix operand order repaired and checked by reading, not compiled, not run on hardware
//
#include <hip/hip_runtime.h>

// ---------------- constants from the reference ----------------
#define NC      21
#define MEMN    1000
#define DIMD    128
#define MARGIN  0.2f
#define K_EASY  40
#define K_HARD  160
#define ANCH    200         // K_EASY + K_HARD
#define NPOS    131072      // 8 * 128 * 128 downsampled positions
#define NROWS   20000       // (NC-1) * MEM memory rows
#define APAD    208         // 13 * 16, padded anchor rows for WMMA

typedef __attribute__((ext_vector_type(16))) _Float16 v16h;
typedef __attribute__((ext_vector_type(8)))  _Float16 v8h;
typedef __attribute__((ext_vector_type(8)))  float    v8f;

// ---------------- small helpers ----------------
__device__ __forceinline__ unsigned f2key(float f) {
    unsigned u = __float_as_uint(f);
    return (u & 0x80000000u) ? ~u : (u | 0x80000000u);
}
__device__ __forceinline__ float key2f(unsigned k) {
    unsigned u = (k & 0x80000000u) ? (k & 0x7FFFFFFFu) : ~k;
    return __uint_as_float(u);
}
__device__ __forceinline__ float wave_sum(float v) {
    for (int o = 16; o > 0; o >>= 1) v += __shfl_xor(v, o, 32);
    return v;
}

// ---------------- K0: init header ----------------
__global__ void init_kernel(int* hdr) {
    if (threadIdx.x == 0) hdr[0] = NC;   // cls accumulator for atomicMin
}

// ---------------- K1: cls = min valid downsampled wsss label ----------------
__global__ void cls_kernel(const int* __restrict__ wsss, int* hdr) {
    int p = blockIdx.x * blockDim.x + threadIdx.x;
    if (p >= NPOS) return;
    int b = p >> 14, rp = (p >> 7) & 127, cp = p & 127;
    long off = ((long)b * 512 + rp * 4) * 512 + cp * 4;
    int sw = wsss[off];
    if (sw != 0 && sw != 255) atomicMin(&hdr[0], sw);
}

// ---------------- K2: ordered pick of first K_EASY easy / K_HARD hard ----------------
__global__ void select_kernel(const int* __restrict__ wsss,
                              const int* __restrict__ fsss,
                              const float* __restrict__ preds,
                              const int* __restrict__ hdr,
                              int* idxArr, int* validArr, int* widxArr) {
    __shared__ int s_easy_base, s_hard_base;
    __shared__ int s_cnt_e[8], s_cnt_h[8];
    __shared__ int s_off_e[8], s_off_h[8];

    const int cls = hdr[0];
    int tid = threadIdx.x, lane = tid & 31, wv = tid >> 5;
    if (tid == 0) { s_easy_base = 0; s_hard_base = 0; }
    __syncthreads();

    for (int chunk = 0; chunk < NPOS; chunk += 256) {
        int p  = chunk + tid;
        int b  = p >> 14, rp = (p >> 7) & 127, cp = p & 127;
        long off = ((long)b * 512 + rp * 4) * 512 + cp * 4;
        int sw = wsss[off];
        int sf = fsss[off];
        bool fg  = (sf != 0) && (sf != 255);
        bool anc = (sw == cls) && fg;
        bool easy = false, hard = false;
        if (anc) {
            // argmax over 21 channels (first max wins)
            long pb = ((long)b * NC * 512 + (long)rp * 4) * 512 + cp * 4;
            float best = preds[pb]; int bi = 0;
            #pragma unroll 4
            for (int ch = 1; ch < NC; ++ch) {
                float v = preds[pb + (long)ch * 262144];
                if (v > best) { best = v; bi = ch; }
            }
            easy = (bi == cls);
            hard = !easy;
        }
        unsigned eb = (unsigned)__ballot(easy);
        unsigned hb = (unsigned)__ballot(hard);
        unsigned lm = (1u << lane) - 1u;
        int ePre = __popc(eb & lm), hPre = __popc(hb & lm);
        if (lane == 0) { s_cnt_e[wv] = __popc(eb); s_cnt_h[wv] = __popc(hb); }
        __syncthreads();
        if (tid == 0) {
            int ae = s_easy_base, ah = s_hard_base;
            for (int w = 0; w < 8; ++w) {
                s_off_e[w] = ae; ae += s_cnt_e[w];
                s_off_h[w] = ah; ah += s_cnt_h[w];
            }
            s_easy_base = ae; s_hard_base = ah;
        }
        __syncthreads();
        if (easy) { int r = s_off_e[wv] + ePre; if (r < K_EASY) idxArr[r] = p; }
        if (hard) { int r = s_off_h[wv] + hPre; if (r < K_HARD) idxArr[K_EASY + r] = p; }
    }
    __syncthreads();
    int e = s_easy_base < K_EASY ? s_easy_base : K_EASY;
    int h = s_hard_base < K_HARD ? s_hard_base : K_HARD;
    if (tid < ANCH) {
        bool v = (tid < K_EASY) ? (tid < e) : ((tid - K_EASY) < h);
        validArr[tid] = v ? 1 : 0;
        widxArr[tid]  = v ? ((tid < K_EASY) ? tid : (e + tid - K_EASY)) : MEMN;
        if (!v) idxArr[tid] = 0;
    }
}

// ---------------- K3: gather + normalize anchors (f32 + f16) ----------------
__global__ void anchor_kernel(const float* __restrict__ emb,
                              const int* __restrict__ idxArr,
                              _Float16* Ah, float* Af) {
    int gw  = (blockIdx.x * blockDim.x + threadIdx.x) >> 5;
    int lane = threadIdx.x & 31;
    if (gw >= APAD) return;
    if (gw < ANCH) {
        int p = idxArr[gw];
        int b = p >> 14, rp = (p >> 7) & 127, cp = p & 127;
        long base = (long)b * 2097152 + (long)rp * 128 + cp;   // + d*16384
        float v[4]; float ss = 0.f;
        #pragma unroll
        for (int t = 0; t < 4; ++t) {
            int d = lane + 32 * t;
            v[t] = emb[base + (long)d * 16384];
            ss += v[t] * v[t];
        }
        ss = wave_sum(ss);
        float inv = 1.0f / fmaxf(sqrtf(ss), 1e-12f);
        #pragma unroll
        for (int t = 0; t < 4; ++t) {
            int d = lane + 32 * t;
            float nv = v[t] * inv;
            Ah[gw * DIMD + d] = (_Float16)nv;
            Af[gw * DIMD + d] = nv;
        }
    } else {
        #pragma unroll
        for (int t = 0; t < 4; ++t) Ah[gw * DIMD + lane + 32 * t] = (_Float16)0.f;
    }
}

// ---------------- K4: normalize memory rows -> f16 ----------------
__global__ void memnorm_kernel(const float* __restrict__ mem, _Float16* Bh) {
    int row  = (blockIdx.x * blockDim.x + threadIdx.x) >> 5;
    int lane = threadIdx.x & 31;
    if (row >= NROWS) return;
    const float* src = mem + (long)row * DIMD;
    float v[4]; float ss = 0.f;
    #pragma unroll
    for (int t = 0; t < 4; ++t) { v[t] = src[lane + 32 * t]; ss += v[t] * v[t]; }
    ss = wave_sum(ss);
    float inv = 1.0f / fmaxf(sqrtf(ss), 1e-8f);
    #pragma unroll
    for (int t = 0; t < 4; ++t) Bh[(long)row * DIMD + lane + 32 * t] = (_Float16)(v[t] * inv);
}

// ---------------- K5: sims = anchors x mem^T via v_wmma_f32_16x16x32_f16 ----------------
// One wave computes a 16x16 output tile, K=128 in four chained WMMAs.
// sims is padded to APAD rows so the epilogue stores are unconditional
// (no exec-mask juggling; consumers only read rows < ANCH).
__global__ void gemm_kernel(const _Float16* __restrict__ Ah,
                            const _Float16* __restrict__ Bh,
                            float* __restrict__ sims) {
    int gw = (blockIdx.x * blockDim.x + threadIdx.x) >> 5;
    const int NT = NROWS / 16;              // 1250 n-tiles
    if (gw >= (APAD / 16) * NT) return;     // 13 * 1250 wave jobs
    int at = gw / NT, nt = gw % NT;
    int lane = threadIdx.x & 31;
    int m  = lane & 15;
    int hi = lane >> 4;

    // ISA 16-bit A 16x32 fragment: lane(m,hi) holds K = hi*8 + j  and  K = 16 + hi*8 + j
    const _Float16* Abase = Ah + (at * 16 + m) * DIMD + hi * 8;
    // ISA 16-bit B 32x16 fragment: lane(n=m,hi) holds K = hi*8 + j  and  K = 16 + hi*8 + j
    const _Float16* Bbase = Bh + (long)(nt * 16 + m) * DIMD + hi * 8;

    v8f c = {};
    #pragma unroll
    for (int kb = 0; kb < DIMD; kb += 32) {
        if (kb + 32 < DIMD) {
            __builtin_prefetch(Abase + kb + 32, 0, 1);
            __builtin_prefetch(Bbase + kb + 32, 0, 1);
        }
        v8h alo = *(const v8h*)(Abase + kb);
        v8h ahi = *(const v8h*)(Abase + kb + 16);
        v16h a;
        #pragma unroll
        for (int j = 0; j < 8; ++j) { a[j] = alo[j]; a[8 + j] = ahi[j]; }
        v8h blo = *(const v8h*)(Bbase + kb);
        v8h bhi = *(const v8h*)(Bbase + kb + 16);
        v16h bfrag;
        #pragma unroll
        for (int j = 0; j < 8; ++j) { bfrag[j] = blo[j]; bfrag[8 + j] = bhi[j]; }
        c = __builtin_amdgcn_wmma_f32_16x16x32_f16(
                /*neg_a=*/false, a, /*neg_b=*/false, bfrag,
                /*c_mod=*/(short)0, c, /*reuse_a=*/false, /*reuse_b=*/false);
    }
    // C/D layout: VGPR r -> row M = hi*8 + r, N = lane&15
    int a0 = at * 16, n0 = nt * 16, n = lane & 15;
    float* dst = sims + (long)(a0 + hi * 8) * NROWS + n0 + n;
    #pragma unroll
    for (int r = 0; r < 8; ++r) dst[(long)r * NROWS] = c[r];
}

// ---------------- K6: per-anchor radix top-1000 + relu pair sum ----------------
__global__ void topk_pair_kernel(const float* __restrict__ sims,
                                 const int* __restrict__ hdr,
                                 float* per_anchor) {
    __shared__ unsigned s_hist[256];
    __shared__ float    s_neg[MEMN];
    __shared__ unsigned s_cnt;
    __shared__ unsigned s_prefix;
    __shared__ int      s_want;
    __shared__ float    s_part[8];

    int a = blockIdx.x;
    int tid = threadIdx.x, lane = tid & 31, wv = tid >> 5;
    int cls = hdr[0]; if (cls < 1) cls = 1; if (cls > NC - 1) cls = NC - 1;
    const float* row = sims + (long)a * NROWS;
    int skip = (cls - 1) * MEMN;

    // 4-pass radix select: key of the 1000th largest negative
    unsigned prefix = 0;
    int want = MEMN;
    for (int shift = 24; shift >= 0; shift -= 8) {
        s_hist[tid] = 0;
        __syncthreads();
        for (int n = tid; n < NROWS; n += 256) {
            if (n >= skip && n < skip + MEMN) continue;
            unsigned key = f2key(row[n]);
            if (shift < 24) {
                unsigned msk = 0xFFFFFFFFu << (shift + 8);
                if ((key & msk) != prefix) continue;
            }
            atomicAdd(&s_hist[(key >> shift) & 0xFF], 1u);
        }
        __syncthreads();
        if (tid == 0) {
            int cum = 0, b = 255;
            for (; b >= 0; --b) {
                int c = (int)s_hist[b];
                if (cum + c >= want) break;
                cum += c;
            }
            if (b < 0) b = 0;
            s_prefix = prefix | ((unsigned)b << shift);
            s_want   = want - cum;
        }
        __syncthreads();
        prefix = s_prefix;
        want   = s_want;
    }
    unsigned Kth = prefix;
    float Tval = key2f(Kth);

    // compact strictly-greater negatives into LDS, pad with threshold value
    if (tid == 0) s_cnt = 0;
    __syncthreads();
    for (int n = tid; n < NROWS; n += 256) {
        if (n >= skip && n < skip + MEMN) continue;
        float v = row[n];
        if (f2key(v) > Kth) {
            unsigned p = atomicAdd(&s_cnt, 1u);
            if (p < MEMN) s_neg[p] = v;
        }
    }
    __syncthreads();
    unsigned cgt = s_cnt > MEMN ? MEMN : s_cnt;
    for (int k = cgt + tid; k < MEMN; k += 256) s_neg[k] = Tval;
    __syncthreads();

    // pair = mean over 1000 pos x 1000 top-neg of relu(pos - neg + margin)
    float acc = 0.f;
    for (int mi = tid; mi < MEMN; mi += 256) {
        float pm = row[skip + mi] + MARGIN;
        float s = 0.f;
        for (int k = 0; k < MEMN; ++k) s += fmaxf(pm - s_neg[k], 0.f);
        acc += s;
    }
    acc = wave_sum(acc);
    if (lane == 0) s_part[wv] = acc;
    __syncthreads();
    if (tid == 0) {
        float t = 0.f;
        for (int w = 0; w < 8; ++w) t += s_part[w];
        per_anchor[a] = t * (1.0f / ((float)MEMN * (float)MEMN));
    }
}

// ---------------- K7: final loss ----------------
__global__ void loss_kernel(const float* __restrict__ per_anchor,
                            const int* __restrict__ validArr,
                            float* out) {
    __shared__ float ps[8], pc[8];
    int tid = threadIdx.x, lane = tid & 31, wv = tid >> 5;
    float s = 0.f, c = 0.f;
    if (tid < ANCH) {
        float v = (float)validArr[tid];
        s = per_anchor[tid] * v;
        c = v;
    }
    s = wave_sum(s); c = wave_sum(c);
    if (lane == 0) { ps[wv] = s; pc[wv] = c; }
    __syncthreads();
    if (tid == 0) {
        float S = 0.f, C = 0.f;
        for (int w = 0; w < 8; ++w) { S += ps[w]; C += pc[w]; }
        out[0] = S / fmaxf(C, 1.0f);
    }
}

// ---------------- K8: copy memory -> out (unaligned dst by 1 float) ----------------
__global__ void copy_kernel(const float* __restrict__ src, float* __restrict__ dst, int n) {
    int i = blockIdx.x * blockDim.x + threadIdx.x;
    if (i < n) dst[i] = src[i];
}

// ---------------- K9: scatter valid anchors into memory class row ----------------
__global__ void memupd_kernel(const float* __restrict__ Af,
                              const int* __restrict__ hdr,
                              const int* __restrict__ validArr,
                              const int* __restrict__ widxArr,
                              float* __restrict__ outMem) {
    int j = blockIdx.x, t = threadIdx.x;
    if (!validArr[j]) return;
    int cls = hdr[0]; if (cls < 1) cls = 1; if (cls > NC - 1) cls = NC - 1;
    int w = widxArr[j];
    if (w >= MEMN) return;
    outMem[((long)(cls - 1) * MEMN + w) * DIMD + t] = Af[j * DIMD + t];
}

// ---------------- host glue ----------------
extern "C" void kernel_launch(void* const* d_in, const int* in_sizes, int n_in,
                              void* d_out, int out_size, void* d_ws, size_t ws_size,
                              hipStream_t stream) {
    const float* preds  = (const float*)d_in[0];
    const float* emb    = (const float*)d_in[1];
    const int*   wsss   = (const int*)d_in[2];
    const int*   fsss   = (const int*)d_in[3];
    const float* memory = (const float*)d_in[4];
    float* out = (float*)d_out;

    char* ws = (char*)d_ws;
    int*      hdr      = (int*)(ws + 0);           // 256 B
    int*      idxArr   = (int*)(ws + 256);         // 1 KB
    int*      validArr = (int*)(ws + 1280);        // 1 KB
    int*      widxArr  = (int*)(ws + 2304);        // 1 KB
    float*    perA     = (float*)(ws + 3328);      // 1 KB
    _Float16* Ah       = (_Float16*)(ws + 4352);   // 208*128*2 = 53248 B
    float*    Af       = (float*)(ws + 57600);     // 200*128*4 = 102400 B
    _Float16* Bh       = (_Float16*)(ws + 160000); // 20000*128*2 = 5.12 MB
    float*    sims     = (float*)(ws + 5280000);   // 208*20000*4 = 16.64 MB (padded)

    init_kernel<<<1, 32, 0, stream>>>(hdr);
    cls_kernel<<<NPOS / 256, 256, 0, stream>>>(wsss, hdr);
    select_kernel<<<1, 256, 0, stream>>>(wsss, fsss, preds, hdr, idxArr, validArr, widxArr);
    anchor_kernel<<<(APAD * 32 + 255) / 256, 256, 0, stream>>>(emb, idxArr, Ah, Af);
    memnorm_kernel<<<(NROWS * 32 + 255) / 256, 256, 0, stream>>>(memory, Bh);
    {
        int jobs = (APAD / 16) * (NROWS / 16);   // 13 * 1250 waves
        gemm_kernel<<<(jobs * 32 + 255) / 256, 256, 0, stream>>>(Ah, Bh, sims);
    }
    topk_pair_kernel<<<ANCH, 256, 0, stream>>>(sims, hdr, perA);
    loss_kernel<<<1, 256, 0, stream>>>(perA, validArr, out);
    {
        int n = (NC - 1) * MEMN * DIMD;          // 2,560,000
        copy_kernel<<<(n + 255) / 256, 256, 0, stream>>>(memory, out + 1, n);
    }
    memupd_kernel<<<ANCH, DIMD, 0, stream>>>(Af, hdr, validArr, widxArr, out + 1);
}
